// Qwen3VLSequentialMoeTextSparseMoeBlock_22522808500779
// MI455X (gfx1250) — matrix operand order repaired and checked by reading, not compiled, not run on hardware
//
#include <hip/hip_runtime.h>
#include <hip/hip_bf16.h>
#include <math.h>

#define H_DIM 2048
#define I_DIM 768
#define E_NUM 16
#define TOPK  8
#define T_NUM 4096
#define AS    40   // LDS row stride in halfwords (80B: 16B aligned, bank-spread)

typedef __attribute__((ext_vector_type(16))) __bf16 v16bf;
typedef __attribute__((ext_vector_type(8)))  __bf16 v8bf;
typedef __attribute__((ext_vector_type(8)))  float  v8f;
typedef __attribute__((ext_vector_type(4)))  int    v4i;

// ---- fp32 -> bf16 (round-to-nearest-even via integer path) ---------------
static __device__ __forceinline__ unsigned short f2bf(float f) {
  unsigned int u = __builtin_bit_cast(unsigned int, f);
  u += 0x7FFFu + ((u >> 16) & 1u);
  return (unsigned short)(u >> 16);
}

static __device__ __forceinline__ unsigned pack2(float lo, float hi) {
  return (unsigned)f2bf(lo) | ((unsigned)f2bf(hi) << 16);
}

static __device__ __forceinline__ uint4 pack8f(float4 q0, float4 q1) {
  uint4 o;
  o.x = pack2(q0.x, q0.y);
  o.y = pack2(q0.z, q0.w);
  o.z = pack2(q1.x, q1.y);
  o.w = pack2(q1.z, q1.w);
  return o;
}

// ---- CDNA5 async global->LDS copy (ASYNCcnt path) -------------------------
#if defined(__has_builtin)
#if __has_builtin(__builtin_amdgcn_global_load_async_to_lds_b128)
#define HAVE_ASYNC_LDS 1
#endif
#endif

#ifdef HAVE_ASYNC_LDS
typedef __attribute__((address_space(1))) v4i gv4i;   // global (prints as __device__)
typedef __attribute__((address_space(3))) v4i lv4i;   // LDS (prints as __shared__)
static __device__ __forceinline__ void async_copy16(const void* g, void* l) {
  __builtin_amdgcn_global_load_async_to_lds_b128((gv4i*)g, (lv4i*)l, 0, 0);
}
static __device__ __forceinline__ void async_wait() {
#if __has_builtin(__builtin_amdgcn_s_wait_asynccnt)
  __builtin_amdgcn_s_wait_asynccnt(0);
#else
  asm volatile("s_wait_asynccnt 0x0" ::: "memory");
#endif
}
#endif

// Copy 16 bf16 (32B) from global to LDS.
static __device__ __forceinline__ void copy_a16(const unsigned short* g,
                                                unsigned short* l) {
#ifdef HAVE_ASYNC_LDS
  async_copy16(g, l);
  async_copy16(g + 8, l + 8);
#else
  const uint4* s = (const uint4*)g;
  uint4 p0 = s[0], p1 = s[1];
  uint4* d = (uint4*)l;
  d[0] = p0; d[1] = p1;
#endif
}

static __device__ __forceinline__ void tile_load_fence() {
#ifdef HAVE_ASYNC_LDS
  async_wait();
#endif
  __syncthreads();
}

// Build a 16-element bf16 fragment from two 16-byte LDS chunks (ds_load_b128 x2).
static __device__ __forceinline__ v16bf frag16(const unsigned short* lo,
                                               const unsigned short* hi) {
  v8bf a = *(const v8bf*)lo;
  v8bf b = *(const v8bf*)hi;
  return __builtin_shufflevector(a, b, 0,1,2,3,4,5,6,7,8,9,10,11,12,13,14,15);
}

// ---------------------------------------------------------------- kernel 1
__global__ void k_convert_x(const float* __restrict__ x,
                            unsigned short* __restrict__ xbf) {
  size_t idx = ((size_t)blockIdx.x * 256 + threadIdx.x) * 8;
  const float4* s = (const float4*)(x + idx);
  *(uint4*)(xbf + idx) = pack8f(s[0], s[1]);
}

// ---------------------------------------------------------------- kernel 2
// One wave (32 lanes) per token: logits, softmax-top8, normalized weights.
__global__ void k_router(const float* __restrict__ x,
                         const float* __restrict__ gw,
                         float* __restrict__ logits_out,
                         float* __restrict__ rw) {
  const int lane = threadIdx.x & 31;
  const int t = blockIdx.x * 8 + (threadIdx.x >> 5);

  float acc[E_NUM];
#pragma unroll
  for (int e = 0; e < E_NUM; ++e) acc[e] = 0.f;

  for (int h = lane; h < H_DIM; h += 32) {
    float xv = x[(size_t)t * H_DIM + h];
#pragma unroll
    for (int e = 0; e < E_NUM; ++e) acc[e] += xv * gw[e * H_DIM + h];
  }
#pragma unroll
  for (int e = 0; e < E_NUM; ++e) {
#pragma unroll
    for (int off = 16; off > 0; off >>= 1)
      acc[e] += __shfl_xor(acc[e], off, 32);
  }
  // every lane now holds all 16 logits
  if (lane < E_NUM) logits_out[(size_t)t * E_NUM + lane] = acc[lane];

  const int e = lane & 15;
  float mx = acc[0];
#pragma unroll
  for (int j = 1; j < E_NUM; ++j) mx = fmaxf(mx, acc[j]);
  int rank = 0;
#pragma unroll
  for (int j = 0; j < E_NUM; ++j)
    rank += (acc[j] > acc[e]) || (acc[j] == acc[e] && j < e);
  const bool sel = rank < TOPK;
  float p = sel ? __expf(acc[e] - mx) : 0.f;
  float s = p;
#pragma unroll
  for (int off = 8; off > 0; off >>= 1) s += __shfl_xor(s, off, 32);
  float w = sel ? p / s : 0.f;
  if (lane < E_NUM) rw[(size_t)t * E_NUM + lane] = w;
}

// ---------------------------------------------------------------- kernel 3
// act[e,t,i] = silu(x @ Wg[e]^T) * (x @ Wu[e]^T), bf16 WMMA, fused epilogue.
__global__ __launch_bounds__(256)
void k_gate_up(const unsigned short* __restrict__ xbf,
               const float* __restrict__ gup,
               unsigned short* __restrict__ act) {
  __shared__ __align__(16) unsigned short Asm[128 * AS];
  __shared__ __align__(16) unsigned short Bgs[64 * AS];
  __shared__ __align__(16) unsigned short Bus[64 * AS];

  const int tid  = threadIdx.x;
  const int lane = tid & 31;
  const int wave = tid >> 5;
  const int wr = wave >> 1, wc = wave & 1;
  const int l16 = lane & 15, half = lane >> 4;
  const int t0 = blockIdx.x * 128;
  const int n0 = blockIdx.y * 64;
  const int e  = blockIdx.z;

  const int arow = tid >> 1, acol = (tid & 1) * 16;   // 128 x 32 A tile
  const int brow = tid >> 2, bcol = (tid & 3) * 8;    // 64 x 32 B tiles

  const unsigned short* aptr = xbf + (size_t)(t0 + arow) * H_DIM + acol;
  const float* gptr = gup + ((size_t)e * 2 * I_DIM + n0 + brow) * H_DIM + bcol;
  const float* uptr = gptr + (size_t)I_DIM * H_DIM;

  v8f accG[2][2] = {};
  v8f accU[2][2] = {};

  for (int k0 = 0; k0 < H_DIM; k0 += 32) {
    copy_a16(aptr + k0, Asm + arow * AS + acol);     // async on CDNA5 path
    { // Bg / Bu tiles: fp32 -> bf16 on the fly
      const float4* sg = (const float4*)(gptr + k0);
      *(uint4*)(Bgs + brow * AS + bcol) = pack8f(sg[0], sg[1]);
      const float4* su = (const float4*)(uptr + k0);
      *(uint4*)(Bus + brow * AS + bcol) = pack8f(su[0], su[1]);
    }
    if (k0 + 32 < H_DIM) {                 // stream weights ahead (global_prefetch_b8)
      __builtin_prefetch(gptr + k0 + 32, 0, 0);
      __builtin_prefetch(uptr + k0 + 32, 0, 0);
    }
    tile_load_fence();

    const unsigned short* ar0 = Asm + (wr * 32 + l16) * AS;
    const unsigned short* ar1 = ar0 + 16 * AS;
    v16bf a0 = frag16(ar0 + 8 * half, ar0 + 16 + 8 * half);
    v16bf a1 = frag16(ar1 + 8 * half, ar1 + 16 + 8 * half);

    const unsigned short* bg0p = Bgs + (wc * 32 + l16) * AS;
    const unsigned short* bg1p = bg0p + 16 * AS;
    v16bf bg0 = frag16(bg0p + 8 * half, bg0p + 16 + 8 * half);
    v16bf bg1 = frag16(bg1p + 8 * half, bg1p + 16 + 8 * half);
    const unsigned short* bu0p = Bus + (wc * 32 + l16) * AS;
    const unsigned short* bu1p = bu0p + 16 * AS;
    v16bf bu0 = frag16(bu0p + 8 * half, bu0p + 16 + 8 * half);
    v16bf bu1 = frag16(bu1p + 8 * half, bu1p + 16 + 8 * half);

    accG[0][0] = __builtin_amdgcn_wmma_f32_16x16x32_bf16(false, a0, false, bg0, (short)0, accG[0][0], false, false);
    accG[0][1] = __builtin_amdgcn_wmma_f32_16x16x32_bf16(false, a0, false, bg1, (short)0, accG[0][1], false, false);
    accG[1][0] = __builtin_amdgcn_wmma_f32_16x16x32_bf16(false, a1, false, bg0, (short)0, accG[1][0], false, false);
    accG[1][1] = __builtin_amdgcn_wmma_f32_16x16x32_bf16(false, a1, false, bg1, (short)0, accG[1][1], false, false);
    accU[0][0] = __builtin_amdgcn_wmma_f32_16x16x32_bf16(false, a0, false, bu0, (short)0, accU[0][0], false, false);
    accU[0][1] = __builtin_amdgcn_wmma_f32_16x16x32_bf16(false, a0, false, bu1, (short)0, accU[0][1], false, false);
    accU[1][0] = __builtin_amdgcn_wmma_f32_16x16x32_bf16(false, a1, false, bu0, (short)0, accU[1][0], false, false);
    accU[1][1] = __builtin_amdgcn_wmma_f32_16x16x32_bf16(false, a1, false, bu1, (short)0, accU[1][1], false, false);
    __syncthreads();
  }

  // silu(g) * u, store bf16
#pragma unroll
  for (int mi = 0; mi < 2; ++mi) {
    const int row = t0 + wr * 32 + mi * 16 + 8 * half;
#pragma unroll
    for (int ni = 0; ni < 2; ++ni) {
      const int col = n0 + wc * 32 + ni * 16 + l16;
#pragma unroll
      for (int r = 0; r < 8; ++r) {
        float g = accG[mi][ni][r];
        float u = accU[mi][ni][r];
        float sv = (g / (1.f + __expf(-g))) * u;
        act[((size_t)e * T_NUM + row + r) * I_DIM + col] = f2bf(sv);
      }
    }
  }
}

// ---------------------------------------------------------------- kernel 4
// out[t,h] = sum_e rw[t,e] * (act[e] @ down_w[e]^T)[t,h] — experts looped
// inside, accumulate in registers, single store. No atomics.
__global__ __launch_bounds__(256)
void k_down_combine(const unsigned short* __restrict__ act,
                    const float* __restrict__ dw,
                    const float* __restrict__ rw,
                    float* __restrict__ out) {
  __shared__ __align__(16) unsigned short Asm[128 * AS];
  __shared__ __align__(16) unsigned short Bsm[64 * AS];

  const int tid  = threadIdx.x;
  const int lane = tid & 31;
  const int wave = tid >> 5;
  const int wr = wave >> 1, wc = wave & 1;
  const int l16 = lane & 15, half = lane >> 4;
  const int t0 = blockIdx.x * 128;
  const int h0 = blockIdx.y * 64;

  const int arow = tid >> 1, acol = (tid & 1) * 16;
  const int brow = tid >> 2, bcol = (tid & 3) * 8;

  v8f accO[2][2] = {};

  for (int e = 0; e < E_NUM; ++e) {
    v8f accT[2][2] = {};
    const unsigned short* aptr =
        act + ((size_t)e * T_NUM + t0 + arow) * I_DIM + acol;
    const float* bptr = dw + ((size_t)e * H_DIM + h0 + brow) * I_DIM + bcol;

    for (int k0 = 0; k0 < I_DIM; k0 += 32) {
      copy_a16(aptr + k0, Asm + arow * AS + acol);   // async on CDNA5 path
      {
        const float4* s = (const float4*)(bptr + k0);
        *(uint4*)(Bsm + brow * AS + bcol) = pack8f(s[0], s[1]);
      }
      if (k0 + 32 < I_DIM) __builtin_prefetch(bptr + k0 + 32, 0, 0);
      tile_load_fence();

      const unsigned short* ar0 = Asm + (wr * 32 + l16) * AS;
      const unsigned short* ar1 = ar0 + 16 * AS;
      v16bf a0 = frag16(ar0 + 8 * half, ar0 + 16 + 8 * half);
      v16bf a1 = frag16(ar1 + 8 * half, ar1 + 16 + 8 * half);
      const unsigned short* b0p = Bsm + (wc * 32 + l16) * AS;
      const unsigned short* b1p = b0p + 16 * AS;
      v16bf b0 = frag16(b0p + 8 * half, b0p + 16 + 8 * half);
      v16bf b1 = frag16(b1p + 8 * half, b1p + 16 + 8 * half);

      accT[0][0] = __builtin_amdgcn_wmma_f32_16x16x32_bf16(false, a0, false, b0, (short)0, accT[0][0], false, false);
      accT[0][1] = __builtin_amdgcn_wmma_f32_16x16x32_bf16(false, a0, false, b1, (short)0, accT[0][1], false, false);
      accT[1][0] = __builtin_amdgcn_wmma_f32_16x16x32_bf16(false, a1, false, b0, (short)0, accT[1][0], false, false);
      accT[1][1] = __builtin_amdgcn_wmma_f32_16x16x32_bf16(false, a1, false, b1, (short)0, accT[1][1], false, false);
      __syncthreads();
    }

    // weight by router prob for this expert, per output row
#pragma unroll
    for (int mi = 0; mi < 2; ++mi) {
      const int row = t0 + wr * 32 + mi * 16 + 8 * half;
      float w[8];
#pragma unroll
      for (int r = 0; r < 8; ++r) w[r] = rw[(size_t)(row + r) * E_NUM + e];
#pragma unroll
      for (int ni = 0; ni < 2; ++ni)
#pragma unroll
        for (int r = 0; r < 8; ++r)
          accO[mi][ni][r] += w[r] * accT[mi][ni][r];
    }
  }

#pragma unroll
  for (int mi = 0; mi < 2; ++mi) {
    const int row = t0 + wr * 32 + mi * 16 + 8 * half;
#pragma unroll
    for (int ni = 0; ni < 2; ++ni) {
      const int col = h0 + wc * 32 + ni * 16 + l16;
#pragma unroll
      for (int r = 0; r < 8; ++r)
        out[(size_t)(row + r) * H_DIM + col] = accO[mi][ni][r];
    }
  }
}

// ---------------------------------------------------------------- launch
extern "C" void kernel_launch(void* const* d_in, const int* in_sizes, int n_in,
                              void* d_out, int out_size, void* d_ws, size_t ws_size,
                              hipStream_t stream) {
  const float* x   = (const float*)d_in[0];   // (B,S,H) fp32
  const float* gw  = (const float*)d_in[1];   // (E,H)
  const float* gup = (const float*)d_in[2];   // (E,2I,H)
  const float* dwn = (const float*)d_in[3];   // (E,H,I)

  float* out    = (float*)d_out;                        // routed_out (T*H)
  float* logits = out + (size_t)T_NUM * H_DIM;          // router_logits (T*E)

  char* ws = (char*)d_ws;
  unsigned short* xbf = (unsigned short*)ws;                              // 16.8 MB
  float* rw = (float*)(ws + (size_t)T_NUM * H_DIM * 2);                   // 256 KB
  unsigned short* act = (unsigned short*)(ws + (size_t)T_NUM * H_DIM * 2
                                             + (size_t)T_NUM * E_NUM * 4); // 100.7 MB

  k_convert_x<<<(T_NUM * H_DIM) / (256 * 8), 256, 0, stream>>>(x, xbf);
  k_router<<<T_NUM / 8, 256, 0, stream>>>(x, gw, logits, rw);

  dim3 gC(T_NUM / 128, I_DIM / 64, E_NUM);
  k_gate_up<<<gC, 256, 0, stream>>>(xbf, gup, act);

  dim3 gD(T_NUM / 128, H_DIM / 64);
  k_down_combine<<<gD, 256, 0, stream>>>(act, dwn, rw, out);
}
